// VQEmbedding_88295937671315
// MI455X (gfx1250) — matrix operand order repaired and checked by reading, not compiled, not run on hardware
//
#include <hip/hip_runtime.h>

typedef __attribute__((ext_vector_type(16))) __bf16 v16bf;
typedef __attribute__((ext_vector_type(8)))  float  v8f;

#define K_CODES   512
#define D_DIM     64
#define HW        4096        // H*W
#define CHW       262144      // D*H*W
#define OUT_PLANE 8388608     // B*D*H*W
#define NUM_TILES 8192        // (B*H*W)/16
#define NTHREADS  256

// One wave handles a 16-pixel tile: 16x512 distance GEMM via bf16 WMMA with
// 3-term bf16 hi/lo emulation of fp32, running argmin in C-fragment layout,
// cross-lane argmin reduction, then coalesced scatter of the winning rows.
__global__ __launch_bounds__(NTHREADS)
void vq_embed_kernel(const float* __restrict__ ze,
                     const float* __restrict__ cb,
                     float* __restrict__ out) {
    // Codebook in WMMA B-fragment order: [tile t (32)][chunk q (2)][lane (32)][j (16)]
    __shared__ __align__(32) __bf16 sB_hi[32768];   // 64 KB
    __shared__ __align__(32) __bf16 sB_lo[32768];   // 64 KB
    __shared__ float sSqr[K_CODES];                 // ||c_k||^2
    __shared__ int   sIdx[NTHREADS / 32][16];       // per-wave argmin scratch

    const int tid = threadIdx.x;

    // ---- one-time per block: convert codebook to bf16 hi/lo B-fragments ----
    for (int i = tid; i < 32768; i += NTHREADS) {
        int j  = i & 15;
        int l  = (i >> 4) & 31;
        int tq = i >> 9;                 // t*2 + q
        int q  = tq & 1, t = tq >> 1;
        int code = t * 16 + (l & 15);                 // B column (lane % 16)
        int d    = q * 32 + (l >> 4) * 16 + j;        // B row (K index)
        float c  = cb[code * D_DIM + d];
        __bf16 hi = (__bf16)c;
        sB_hi[i] = hi;
        sB_lo[i] = (__bf16)(c - (float)hi);
    }
    for (int k = tid; k < K_CODES; k += NTHREADS) {
        float s = 0.f;
        #pragma unroll 8
        for (int d = 0; d < D_DIM; ++d) { float c = cb[k * D_DIM + d]; s += c * c; }
        sSqr[k] = s;
    }
    __syncthreads();

    const int lane = tid & 31;
    const int wave = tid >> 5;
    const int col  = lane & 15;      // N index within 16 / pixel for A rows
    const int half = lane >> 4;      // 0: lanes 0-15, 1: lanes 16-31

    const int waveGlobal = blockIdx.x * (NTHREADS / 32) + wave;
    const int totalWaves = gridDim.x * (NTHREADS / 32);

    for (int tile = waveGlobal; tile < NUM_TILES; tile += totalWaves) {
        const int n0  = tile * 16;           // 16 consecutive pixels: same b,h
        const int b   = n0 >> 12;
        const int rem = n0 & 4095;
        const int h   = rem >> 6;
        const int w0  = rem & 63;
        const int basex = b * CHW + h * 64 + w0 + col;   // this lane's pixel (M=col)

        // ---- load A fragments (16-bit A 16x32 layout), bf16 hi/lo split ----
        v16bf a_hi[2], a_lo[2];
        #pragma unroll
        for (int q = 0; q < 2; ++q) {
            #pragma unroll
            for (int j = 0; j < 16; ++j) {
                int d = q * 32 + half * 16 + j;
                float x = ze[basex + d * HW];
                __bf16 hx = (__bf16)x;
                a_hi[q][j] = hx;
                a_lo[q][j] = (__bf16)(x - (float)hx);
            }
        }

        float minv[8];
        int   mini[8];
        #pragma unroll
        for (int r = 0; r < 8; ++r) { minv[r] = 3.4e38f; mini[r] = 0x7fffffff; }

        // ---- 32 code tiles: 6 bf16 WMMAs each (hi*hi + lo*hi + hi*lo) ----
        #pragma unroll 4
        for (int t = 0; t < 32; ++t) {
            const v16bf bh0 = *reinterpret_cast<const v16bf*>(sB_hi + ((t * 2 + 0) * 32 + lane) * 16);
            const v16bf bh1 = *reinterpret_cast<const v16bf*>(sB_hi + ((t * 2 + 1) * 32 + lane) * 16);
            const v16bf bl0 = *reinterpret_cast<const v16bf*>(sB_lo + ((t * 2 + 0) * 32 + lane) * 16);
            const v16bf bl1 = *reinterpret_cast<const v16bf*>(sB_lo + ((t * 2 + 1) * 32 + lane) * 16);
            v8f acc = {};
            acc = __builtin_amdgcn_wmma_f32_16x16x32_bf16(false, a_hi[0], false, bh0, (short)0, acc, false, false);
            acc = __builtin_amdgcn_wmma_f32_16x16x32_bf16(false, a_hi[1], false, bh1, (short)0, acc, false, false);
            acc = __builtin_amdgcn_wmma_f32_16x16x32_bf16(false, a_lo[0], false, bh0, (short)0, acc, false, false);
            acc = __builtin_amdgcn_wmma_f32_16x16x32_bf16(false, a_lo[1], false, bh1, (short)0, acc, false, false);
            acc = __builtin_amdgcn_wmma_f32_16x16x32_bf16(false, a_hi[0], false, bl0, (short)0, acc, false, false);
            acc = __builtin_amdgcn_wmma_f32_16x16x32_bf16(false, a_hi[1], false, bl1, (short)0, acc, false, false);

            const int code = t * 16 + col;       // this lane's code column
            const float csq = sSqr[code];
            #pragma unroll
            for (int r = 0; r < 8; ++r) {
                float dist = csq - 2.0f * acc[r];     // ||c||^2 - 2 x.c
                if (dist < minv[r]) { minv[r] = dist; mini[r] = code; }
            }
        }

        // ---- argmin across the 16 lanes of each half (lexicographic) ----
        #pragma unroll
        for (int off = 1; off < 16; off <<= 1) {
            #pragma unroll
            for (int r = 0; r < 8; ++r) {
                float ov = __shfl_xor(minv[r], off, 32);
                int   oi = __shfl_xor(mini[r], off, 32);
                if (ov < minv[r] || (ov == minv[r] && oi < mini[r])) { minv[r] = ov; mini[r] = oi; }
            }
        }
        if (col == 0) {
            #pragma unroll
            for (int r = 0; r < 8; ++r) sIdx[wave][half * 8 + r] = mini[r];
        }
        // wave-internal LDS RAW: DS ops are in-order per wave; compiler waits DScnt.

        // ---- scatter winning f32 codebook rows; lanes 16-31 write copy #2 ----
        const int p = col;                          // pixel within tile
        const int myIdx = sIdx[wave][p];
        const float* __restrict__ crow = cb + myIdx * D_DIM;
        const int baseo = half * OUT_PLANE + b * CHW + h * 64 + w0 + p;
        #pragma unroll
        for (int d4 = 0; d4 < 16; ++d4) {
            float4 v = reinterpret_cast<const float4*>(crow)[d4];
            out[baseo + (d4 * 4 + 0) * HW] = v.x;
            out[baseo + (d4 * 4 + 1) * HW] = v.y;
            out[baseo + (d4 * 4 + 2) * HW] = v.z;
            out[baseo + (d4 * 4 + 3) * HW] = v.w;
        }
    }
}

extern "C" void kernel_launch(void* const* d_in, const int* in_sizes, int n_in,
                              void* d_out, int out_size, void* d_ws, size_t ws_size,
                              hipStream_t stream) {
    const float* ze = (const float*)d_in[0];   // [32,64,64,64] f32
    const float* cb = (const float*)d_in[1];   // [512,64] f32
    float* out = (float*)d_out;                // 2 x [32,64,64,64] f32
    (void)in_sizes; (void)n_in; (void)out_size; (void)d_ws; (void)ws_size;
    // 512 blocks x 8 waves = 4096 waves -> 2 tiles each; ~130KB LDS -> 2 blocks/WGP.
    vq_embed_kernel<<<512, NTHREADS, 0, stream>>>(ze, cb, out);
}
